// MultiHeadAttentionBlock_850403524885
// MI455X (gfx1250) — compile-verified, matrix operand order checked by reading, not run on hardware
//
#include <hip/hip_runtime.h>

// ---------------------------------------------------------------------------
// MI455X (gfx1250) fused multi-head attention block.
//   B=4, S=2048, D=1024, H=16, DK=64
//   out = MHA(q,k,v; causal) with Q/K/V/O projections.
// Strategy: bf16 WMMA (v_wmma_f32_16x16x32_bf16) everywhere, flash-attention
// style fusion so scores never hit HBM, bf16 intermediates in workspace.
// CDNA5 paths: v_wmma_*, global_load_async_to_lds_b128 (ASYNCcnt),
// global_prefetch, wave32 cross-lane reductions.
// ---------------------------------------------------------------------------

typedef __attribute__((ext_vector_type(16))) __bf16 v16bf;
typedef __attribute__((ext_vector_type(8)))  float  v8f;

union FragU {
    uint4 u4[2];
    v16bf v;
};
union U16x8 {
    uint4 u;
    unsigned short s[8];
};

__device__ __forceinline__ unsigned short bf16_rne(float x) {
    unsigned int u = __float_as_uint(x);
    u += 0x7fffu + ((u >> 16) & 1u);
    return (unsigned short)(u >> 16);
}

// Build a 16xbf16 fragment from two 16-byte runs (ISA layout: each lane holds
// two contiguous 8-element K-runs; lanes 0-15 take runs at +0/+16, lanes
// 16-31 at +8/+24 -- the caller passes the right base pointers).
__device__ __forceinline__ v16bf load_frag16(const unsigned short* p0,
                                             const unsigned short* p1) {
    FragU f;
    f.u4[0] = *(const uint4*)p0;
    f.u4[1] = *(const uint4*)p1;
    return f.v;
}

__device__ __forceinline__ v8f wmma_bf16(v16bf a, v16bf b, v8f c) {
    // 8 args: (neg_a, A, neg_b, B, c_mod, C, reuse_a, reuse_b)
    return __builtin_amdgcn_wmma_f32_16x16x32_bf16(
        false, a, false, b, (short)0, c, false, false);
}

__device__ __forceinline__ float rmax16(float v) {
    v = fmaxf(v, __shfl_xor(v, 8, 32));
    v = fmaxf(v, __shfl_xor(v, 4, 32));
    v = fmaxf(v, __shfl_xor(v, 2, 32));
    v = fmaxf(v, __shfl_xor(v, 1, 32));
    return v;
}
__device__ __forceinline__ float rsum16(float v) {
    v += __shfl_xor(v, 8, 32);
    v += __shfl_xor(v, 4, 32);
    v += __shfl_xor(v, 2, 32);
    v += __shfl_xor(v, 1, 32);
    return v;
}

// Async copy of 16 bytes global -> LDS (GLOBAL_LOAD_ASYNC_TO_LDS_B128,
// tracked by ASYNCcnt). INST_OFFSET is added to BOTH addresses per the ISA.
__device__ __forceinline__ void async_copy_b128(unsigned lds_addr,
                                                unsigned long long gaddr) {
    asm volatile("global_load_async_to_lds_b128 %0, %1, off"
                 :: "v"(lds_addr), "v"(gaddr) : "memory");
}
__device__ __forceinline__ void async_copy_b128_off16(unsigned lds_addr,
                                                      unsigned long long gaddr) {
    asm volatile("global_load_async_to_lds_b128 %0, %1, off offset:16"
                 :: "v"(lds_addr), "v"(gaddr) : "memory");
}
__device__ __forceinline__ void wait_asynccnt0() {
#if defined(__has_builtin)
#if __has_builtin(__builtin_amdgcn_s_wait_asynccnt)
    __builtin_amdgcn_s_wait_asynccnt(0);
    return;
#endif
#endif
    asm volatile("s_wait_asynccnt 0x0" ::: "memory");
}

// ---------------------------------------------------------------------------
// GEMM + bias:  Out[M,N] = A[M,K] @ W[K,N] + bias[N]
//   A_IS_BF16=0: A is f32 (raw q/k/v input), converted to bf16 in LDS.
//   A_IS_BF16=1: A is bf16 row-major (attention context); staged to LDS with
//                async-to-LDS b128 copies (no conversion needed).
//   OUT_F32=1  : write f32 row-major [M,N] (final output).
//   OUT_F32=0  : write bf16 head-major [B,H,S,DK] (projections).
// Block = 256 thr (8 waves), tile 128x128, K-step 32.
// Wave grid 4(M) x 2(N): each wave owns 32x64 = 2x4 WMMA accumulators.
// ---------------------------------------------------------------------------
template <bool A_IS_BF16, bool OUT_F32>
__global__ __launch_bounds__(256)
void gemm_bias_kernel(const void* __restrict__ Aptr,
                      const float* __restrict__ W,
                      const float* __restrict__ bias,
                      void* __restrict__ Out,
                      int M, int N, int K, int S, int H, int DK) {
    __shared__ __align__(16) unsigned short lA[128 * 32];  // [m][k] bf16
    __shared__ __align__(16) unsigned short lB[128 * 32];  // [n][k] bf16 (W^T tile)

    const int tid  = threadIdx.x;
    const int lane = tid & 31;
    const int wid  = tid >> 5;
    const int lh   = lane & 15;
    const int c0   = (lane < 16) ? 0 : 8;

    const int m0 = blockIdx.y * 128;
    const int n0 = blockIdx.x * 128;
    const int wm = (wid & 3) * 32;
    const int wn = (wid >> 2) * 64;

    v8f acc[2][4];
#pragma unroll
    for (int i = 0; i < 2; ++i)
#pragma unroll
        for (int j = 0; j < 4; ++j)
            acc[i][j] = (v8f){0.f, 0.f, 0.f, 0.f, 0.f, 0.f, 0.f, 0.f};

    // staging assignments
    const int a_row = tid >> 1;            // 0..127
    const int a_col = (tid & 1) * 16;      // 0 or 16
    const int b_kr  = tid >> 3;            // 0..31
    const int b_ns  = (tid & 7) * 16;      // 0..112

    for (int k0 = 0; k0 < K; k0 += 32) {
        // ---- stage A tile ----
        if (A_IS_BF16) {
            // pure byte-copy: use async global->LDS (ASYNCcnt path)
            const unsigned short* A = (const unsigned short*)Aptr;
            const unsigned short* src = A + (size_t)(m0 + a_row) * K + k0 + a_col;
            const unsigned lds0 = (unsigned)(size_t)&lA[a_row * 32 + a_col];
            const unsigned long long g0 = (unsigned long long)(size_t)src;
            async_copy_b128(lds0, g0);
            async_copy_b128_off16(lds0, g0);
            if (k0 + 32 < K) __builtin_prefetch(src + 32, 0, 3);
        } else {
            const float* A = (const float*)Aptr;
            const float* src = A + (size_t)(m0 + a_row) * K + k0 + a_col;
#pragma unroll
            for (int j = 0; j < 16; j += 4) {
                float4 f = *(const float4*)(src + j);
                unsigned short* d = &lA[a_row * 32 + a_col + j];
                d[0] = bf16_rne(f.x); d[1] = bf16_rne(f.y);
                d[2] = bf16_rne(f.z); d[3] = bf16_rne(f.w);
            }
            if (k0 + 32 < K) __builtin_prefetch(src + 32, 0, 3);
        }
        // ---- stage B tile transposed: lB[n][k] = W[k0+k][n0+n] ----
        {
            const float* src = W + (size_t)(k0 + b_kr) * N + n0 + b_ns;
#pragma unroll
            for (int j = 0; j < 16; j += 4) {
                float4 f = *(const float4*)(src + j);
                lB[(b_ns + j + 0) * 32 + b_kr] = bf16_rne(f.x);
                lB[(b_ns + j + 1) * 32 + b_kr] = bf16_rne(f.y);
                lB[(b_ns + j + 2) * 32 + b_kr] = bf16_rne(f.z);
                lB[(b_ns + j + 3) * 32 + b_kr] = bf16_rne(f.w);
            }
            if (k0 + 32 < K) __builtin_prefetch(src + (size_t)32 * N, 0, 3);
        }
        if (A_IS_BF16) wait_asynccnt0();
        __syncthreads();

        // ---- fragments + WMMA ----
        v16bf afrag[2], bfrag[4];
#pragma unroll
        for (int ms = 0; ms < 2; ++ms) {
            const unsigned short* p = &lA[(wm + ms * 16 + lh) * 32];
            afrag[ms] = load_frag16(p + c0, p + 16 + c0);
        }
#pragma unroll
        for (int ns = 0; ns < 4; ++ns) {
            const unsigned short* p = &lB[(wn + ns * 16 + lh) * 32];
            bfrag[ns] = load_frag16(p + c0, p + 16 + c0);
        }
#pragma unroll
        for (int ms = 0; ms < 2; ++ms)
#pragma unroll
            for (int ns = 0; ns < 4; ++ns)
                acc[ms][ns] = wmma_bf16(afrag[ms], bfrag[ns], acc[ms][ns]);
        __syncthreads();
    }

    // ---- epilogue ----
#pragma unroll
    for (int ms = 0; ms < 2; ++ms) {
#pragma unroll
        for (int ns = 0; ns < 4; ++ns) {
            const int ng = n0 + wn + ns * 16 + lh;
            const float bv = bias[ng];
            const int mbase = m0 + wm + ms * 16 + ((lane < 16) ? 0 : 8);
#pragma unroll
            for (int r = 0; r < 8; ++r) {
                const int mg = mbase + r;
                const float v = acc[ms][ns][r] + bv;
                if (OUT_F32) {
                    ((float*)Out)[(size_t)mg * N + ng] = v;
                } else {
                    // bf16 head-major: [B,H,S,DK]
                    const int bidx = mg / S, s = mg - bidx * S;
                    const int h = ng / DK, d = ng - h * DK;
                    ((unsigned short*)Out)[(((size_t)bidx * H + h) * S + s) * DK + d] =
                        bf16_rne(v);
                }
            }
        }
    }
}

// ---------------------------------------------------------------------------
// Flash attention (causal), bf16 in / bf16 out.
//   Qp/Kp/Vp: [B,H,S,64] bf16.  Ctx: [B,S,D] bf16 (heads merged).
// Block = 256 thr (8 waves), Q tile 128 rows (16 rows / wave), KV step 32.
// Per KV step & wave: scores = 2x(16x16) C-frags from 4 WMMAs, online
// softmax via cross-lane reductions, P restaged via per-wave LDS, then
// 4 WMMAs accumulate the 16x64 O tile.
// ---------------------------------------------------------------------------
__global__ __launch_bounds__(256)
void flash_attn_kernel(const unsigned short* __restrict__ Qp,
                       const unsigned short* __restrict__ Kp,
                       const unsigned short* __restrict__ Vp,
                       unsigned short* __restrict__ Ctx,
                       int S, int H, int D) {
    __shared__ __align__(16) unsigned short lV[64 * 32];       // [d][kv] V^T chunk
    __shared__ __align__(16) unsigned short lP[8 * 16 * 32];   // per-wave P tile

    const int tid  = threadIdx.x;
    const int lane = tid & 31;
    const int wid  = tid >> 5;
    const int lh   = lane & 15;
    const int c0   = (lane < 16) ? 0 : 8;

    const int b = blockIdx.z, h = blockIdx.y;
    const size_t bh = (size_t)b * H + h;
    const int q0 = blockIdx.x * 128;

    // Q fragments (A-matrix rows: lane lh -> Q row)
    const int qrowA = q0 + wid * 16 + lh;
    const unsigned short* qb = Qp + (bh * S + qrowA) * 64;
    v16bf aQ[2];
    aQ[0] = load_frag16(qb + c0,      qb + 16 + c0);
    aQ[1] = load_frag16(qb + 32 + c0, qb + 48 + c0);

    v8f o[4];
#pragma unroll
    for (int f = 0; f < 4; ++f)
        o[f] = (v8f){0.f, 0.f, 0.f, 0.f, 0.f, 0.f, 0.f, 0.f};
    float mrow[8], lrow[8];
#pragma unroll
    for (int r = 0; r < 8; ++r) { mrow[r] = -1e30f; lrow[r] = 0.f; }

    const int rbase = q0 + wid * 16 + ((lane < 16) ? 0 : 8);  // C-frag row base
    const int kv_hi = q0 + 128;  // causal upper bound for this Q tile (<= S)
    const int v_kv = tid >> 3;         // 0..31
    const int v_ds = (tid & 7) * 8;    // 0..56

    for (int kv0 = 0; kv0 < kv_hi; kv0 += 32) {
        __syncthreads();
        // stage V chunk transposed: lV[d][kv]
        {
            U16x8 t;
            t.u = *(const uint4*)(Vp + (bh * S + kv0 + v_kv) * 64 + v_ds);
#pragma unroll
            for (int j = 0; j < 8; ++j)
                lV[(v_ds + j) * 32 + v_kv] = t.s[j];
        }
        __syncthreads();

        // ---- scores: Q(16x64) @ K^T(64x32) ----
        v8f sc[2];
#pragma unroll
        for (int ns = 0; ns < 2; ++ns) {
            const int kvc = kv0 + ns * 16 + lh;  // B-matrix col = KV row
            const unsigned short* kb = Kp + (bh * S + kvc) * 64;
            v8f c = (v8f){0.f, 0.f, 0.f, 0.f, 0.f, 0.f, 0.f, 0.f};
            c = wmma_bf16(aQ[0], load_frag16(kb + c0,      kb + 16 + c0), c);
            c = wmma_bf16(aQ[1], load_frag16(kb + 32 + c0, kb + 48 + c0), c);
            sc[ns] = c;
        }

        // ---- online softmax + P staging ----
        unsigned short* pw = &lP[wid * 512];
#pragma unroll
        for (int r = 0; r < 8; ++r) {
            const int qg = rbase + r;
            float s0 = sc[0][r] * 0.125f;  // 1/sqrt(64)
            float s1 = sc[1][r] * 0.125f;
            if (kv0 + lh      > qg) s0 = -1e9f;
            if (kv0 + 16 + lh > qg) s1 = -1e9f;
            const float mx = rmax16(fmaxf(s0, s1));
            const float nm = fmaxf(mrow[r], mx);
            const float alpha = __expf(mrow[r] - nm);
            mrow[r] = nm;
            const float p0 = __expf(s0 - nm);
            const float p1 = __expf(s1 - nm);
            lrow[r] = lrow[r] * alpha + rsum16(p0 + p1);
#pragma unroll
            for (int f = 0; f < 4; ++f) o[f][r] *= alpha;
            const int pr = r + ((lane < 16) ? 0 : 8);
            pw[pr * 32 + lh]      = bf16_rne(p0);
            pw[pr * 32 + 16 + lh] = bf16_rne(p1);
        }

        // ---- O += P(16x32) @ V(32x64) ----
        const unsigned short* pb = &pw[lh * 32];
        const v16bf aP = load_frag16(pb + c0, pb + 16 + c0);
#pragma unroll
        for (int f = 0; f < 4; ++f) {
            const unsigned short* vb = &lV[(f * 16 + lh) * 32];
            o[f] = wmma_bf16(aP, load_frag16(vb + c0, vb + 16 + c0), o[f]);
        }
    }

    // ---- normalize + store merged-head bf16 ctx ----
#pragma unroll
    for (int f = 0; f < 4; ++f) {
        const int d = h * 64 + f * 16 + lh;
#pragma unroll
        for (int r = 0; r < 8; ++r) {
            const int qg = rbase + r;
            const float v = o[f][r] / lrow[r];
            Ctx[((size_t)b * S + qg) * D + d] = bf16_rne(v);
        }
    }
}

// ---------------------------------------------------------------------------
extern "C" void kernel_launch(void* const* d_in, const int* in_sizes, int n_in,
                              void* d_out, int out_size, void* d_ws, size_t ws_size,
                              hipStream_t stream) {
    constexpr int B = 4, S = 2048, D = 1024, H = 16, DK = 64;
    constexpr int M = B * S;  // 8192

    const float* q   = (const float*)d_in[0];
    const float* k   = (const float*)d_in[1];
    const float* v   = (const float*)d_in[2];
    // d_in[3] is the causal mask -- implemented analytically in-kernel.
    const float* w_q = (const float*)d_in[4];
    const float* b_q = (const float*)d_in[5];
    const float* w_k = (const float*)d_in[6];
    const float* b_k = (const float*)d_in[7];
    const float* w_v = (const float*)d_in[8];
    const float* b_v = (const float*)d_in[9];
    const float* w_o = (const float*)d_in[10];
    const float* b_o = (const float*)d_in[11];

    const size_t projElems = (size_t)B * H * S * DK;  // == B*S*D
    unsigned short* Qp  = (unsigned short*)d_ws;
    unsigned short* Kp  = Qp + projElems;
    unsigned short* Vp  = Kp + projElems;
    unsigned short* Ctx = Vp + projElems;

    dim3 blk(256);
    dim3 gg(D / 128, M / 128);  // (8, 64)

    // projections -> bf16 head-major [B,H,S,DK]
    gemm_bias_kernel<false, false><<<gg, blk, 0, stream>>>(q, w_q, b_q, Qp, M, D, D, S, H, DK);
    gemm_bias_kernel<false, false><<<gg, blk, 0, stream>>>(k, w_k, b_k, Kp, M, D, D, S, H, DK);
    gemm_bias_kernel<false, false><<<gg, blk, 0, stream>>>(v, w_v, b_v, Vp, M, D, D, S, H, DK);

    // fused causal attention -> bf16 [B,S,D]
    dim3 ag(S / 128, H, B);  // (16, 16, 4)
    flash_attn_kernel<<<ag, blk, 0, stream>>>(Qp, Kp, Vp, Ctx, S, H, D);

    // output projection -> f32 [B,S,D]
    gemm_bias_kernel<true, true><<<gg, blk, 0, stream>>>(Ctx, w_o, b_o, (float*)d_out, M, D, D, S, H, DK);
}
